// MinMajThinCubeSimulatorKeOps_67121748902102
// MI455X (gfx1250) — compile-verified, matrix operand order checked by reading, not run on hardware
//
#include <hip/hip_runtime.h>

typedef __attribute__((ext_vector_type(16))) _Float16 v16h;
typedef __attribute__((ext_vector_type(8)))  float    v8f;

#define DOUT   64          // output depth channels
#define JQ     256         // velocity-quantization grid (K of the GEMM)
#define HOUT   128
#define WOUT   128
#define NPIX   (HOUT*WOUT) // 16384 output pixels (N of the GEMM)
#define PADC   108         // int(0.5*(sqrt(2)-1)*512)+2
#define HP     728         // 512 + 2*PAD
#define WP     472         // 256 + 2*PAD
#define HFINE  512
#define WFINE  256

// C_KMS * (F_REST - f_fine[d]) / F_REST, computed in double like numpy, cast to f32
__device__ __forceinline__ float vlab_fine(int d, float vshift) {
  double f_off = 63.75e6 - 0.5e6 * (double)d;    // F_REST - f_fine[d]
  float v = (float)(299792.458 * f_off / 230538000000.0);
  return v - vshift;
}

// ---------------------------------------------------------------------------
// Kernel 1: spectral dictionary  Tavg[do][j] = (1/32) * sum_f exp(k*(vl[4do+f]-vq_j)^2)
// 64 x 256 f16 table -> d_ws (32 KB). Only 65536 exps total.
// ---------------------------------------------------------------------------
__global__ __launch_bounds__(256) void k_spectral_table(
    const float* __restrict__ p_incl, const float* __restrict__ p_lb,
    const float* __restrict__ p_vshift, const float* __restrict__ p_vmax,
    _Float16* __restrict__ T)
{
  int idx = blockIdx.x * 256 + threadIdx.x;
  if (idx >= DOUT * JQ) return;
  int j  = idx & (JQ - 1);
  int dO = idx >> 8;

  float lb    = *p_lb;
  float kcoef = -0.5f / (lb * lb);
  float vm    = fmaxf(fabsf((*p_vmax) * sinf(*p_incl)), 1e-6f); // |v_los| < vm strictly
  float hstep = 2.0f * vm / (float)(JQ - 1);
  float vq    = -vm + hstep * (float)j;
  float vs    = *p_vshift;

  float acc = 0.0f;
#pragma unroll
  for (int f = 0; f < 4; ++f) {
    float vl = vlab_fine(4 * dO + f, vs);
    float d  = vl - vq;
    acc += __expf(kcoef * d * d);
  }
  T[dO * JQ + j] = (_Float16)(acc * 0.03125f);   // folds the full 1/32 block mean
}

// ---------------------------------------------------------------------------
// Kernel 2: per 32-pixel block, scatter 32 taps/pixel into LDS W[j][px] (f32
// atomics), then WMMA GEMM  out(64 x 32) = Tavg(64 x 256) * W(256 x 32).
// 8 wave32 per WG: 4 do-tiles x 2 px-tiles of 16x16, K=256 in 8 wmma steps.
// ---------------------------------------------------------------------------
__global__ __launch_bounds__(256) void k_main(
    const float* __restrict__ p_incl, const float* __restrict__ p_rot,
    const float* __restrict__ p_lb,   const float* __restrict__ p_vmax,
    const float* __restrict__ p_rturn,const float* __restrict__ p_i0,
    const float* __restrict__ p_rd,
    const _Float16* __restrict__ Tg, float* __restrict__ out)
{
  __shared__ __align__(16) _Float16 Tl[DOUT * JQ]; // 32 KB dictionary
  __shared__ __align__(16) float    Wl[JQ * 32];   // 32 KB coefficient tile

  const int tid = threadIdx.x;

  // Stage dictionary (2048 float4) and zero W (8192 floats)
  {
    const float4* src = (const float4*)Tg;
    float4*       dst = (float4*)Tl;
#pragma unroll
    for (int i = 0; i < 8; ++i) dst[tid + 256 * i] = src[tid + 256 * i];
  }
#pragma unroll
  for (int i = 0; i < 32; ++i) Wl[tid + 256 * i] = 0.0f;
  __syncthreads();

  // Scalars (uniform)
  float incl = *p_incl, rot = *p_rot, lb = *p_lb;
  float vmax = *p_vmax, rturn = *p_rturn, i0v = *p_i0, rd = *p_rd;
  float cos_i = cosf(incl), sin_i = sinf(incl);
  float ca = cosf(-rot), sa = sinf(-rot);
  const float kap = 0.5f * 727.0f / 471.0f;        // (U_Y/U)*(HP-1)/(WP-1)
  float norm = 0.3989422804014327f / lb;           // 1/sqrt(2*pi*sig2)
  float vm   = fmaxf(fabsf(vmax * sin_i), 1e-6f);  // same expr as kernel 1
  float invh = (float)(JQ - 1) / (2.0f * vm);
  float v0   = -vm;

  const int pxbase = blockIdx.x * 32;

  // 32 px * 32 taps = 1024 taps, 4 per thread
#pragma unroll
  for (int i = 0; i < 4; ++i) {
    int g   = tid + 256 * i;
    int pxl = g >> 5;          // pixel within block [0,32)
    int t   = g & 31;          // tap id: 8 fine positions x 4 bilinear corners
    int px  = pxbase + pxl;
    int ho  = px >> 7, wo = px & 127;
    int s   = t >> 2;
    int uh  = s >> 1, uw = s & 1;
    int b   = t & 3,  bx = b & 1, by = b >> 1;

    int hp = ho * 4 + uh + PADC;                  // padded row coord
    int wp = wo * 2 + uw + PADC;                  // padded col coord
    float xs = -1.0f + 2.0f * (float)wp / (float)(WP - 1);
    float ys = -1.0f + 2.0f * (float)hp / (float)(HP - 1);
    float gx = ca * xs - sa * kap * ys;
    float gy = (sa / kap) * xs + ca * ys;
    float ix = fminf(fmaxf((gx + 1.0f) * 0.5f * (float)(WP - 1), 0.0f), (float)(WP - 1));
    float iy = fminf(fmaxf((gy + 1.0f) * 0.5f * (float)(HP - 1), 0.0f), (float)(HP - 1));
    float ix0f = floorf(ix), iy0f = floorf(iy);
    float wx = ix - ix0f,    wy = iy - iy0f;
    int ix0 = (int)ix0f, iy0 = (int)iy0f;
    int ix1 = min(ix0 + 1, WP - 1), iy1 = min(iy0 + 1, HP - 1);
    int ixc = bx ? ix1 : ix0;
    int iyc = by ? iy1 : iy0;
    float wgt = (bx ? wx : 1.0f - wx) * (by ? wy : 1.0f - wy);

    int hs = iyc - PADC, ws = ixc - PADC;         // unpadded source coords
    if (hs >= 0 && hs < HFINE && ws >= 0 && ws < WFINE && wgt > 0.0f) {
      float x = -5.0f + 0.01953125f * ((float)hs + 0.5f);             // _X[hs]
      float y = (-5.0f + 0.0390625f * ((float)ws + 0.5f)) / cos_i;    // _Y[ws]/cos_i
      float R = sqrtf(x * x + y * y + 1e-12f);
      float Ipix = i0v * __expf(-R / rd);
      float vrot = vmax * 0.6366197723675814f * atanf(R / rturn);     // 2/pi
      float vlos = vrot * sin_i * (x / R);
      float c    = wgt * Ipix * norm;

      // hat interpolation onto the velocity grid
      float jf = (vlos - v0) * invh;
      jf = fminf(fmaxf(jf, 0.0f), (float)(JQ - 1) - 0.001f);
      int   j0 = (int)jf;
      float af = jf - (float)j0;
      atomicAdd(&Wl[j0 * 32 + pxl],       c * (1.0f - af));
      atomicAdd(&Wl[(j0 + 1) * 32 + pxl], c * af);
    }
  }
  __syncthreads();

  // ---- WMMA GEMM: each wave owns a 16(do) x 16(px) tile, K = 256 ----
  const int wave = tid >> 5, lane = tid & 31;
  const int halfid = lane >> 4, mn = lane & 15;
  const int dotile = wave >> 1, pxtile = wave & 1;
  const _Float16* Arow = &Tl[(dotile * 16 + mn) * JQ];  // A row = output depth
  const int col = pxtile * 16 + mn;                      // B col = pixel

  v8f acc = {0.f, 0.f, 0.f, 0.f, 0.f, 0.f, 0.f, 0.f};
#pragma unroll
  for (int kb = 0; kb < JQ; kb += 32) {
    // A fragment, 16-bit 16x32 layout: elems 0..7 -> K = kb+8*half+e,
    //                                  elems 8..15 -> K = kb+16+8*half+e
    v16h a, bf;
#pragma unroll
    for (int e = 0; e < 8; ++e) a[e]     = Arow[kb + 8 * halfid + e];
#pragma unroll
    for (int e = 0; e < 8; ++e) a[8 + e] = Arow[kb + 16 + 8 * halfid + e];
    // B fragment (K-major per lane on column 'col'), convert f32 -> f16
#pragma unroll
    for (int e = 0; e < 8; ++e) bf[e]     = (_Float16)Wl[(kb + 8 * halfid + e) * 32 + col];
#pragma unroll
    for (int e = 0; e < 8; ++e) bf[8 + e] = (_Float16)Wl[(kb + 16 + 8 * halfid + e) * 32 + col];

    acc = __builtin_amdgcn_wmma_f32_16x16x32_f16(
        /*neg_a=*/false, a, /*neg_b=*/false, bf,
        /*c_mod=*/(short)0, acc, /*reuse_a=*/false, /*reuse_b=*/false);
  }

  // C/D layout: VGPR r -> M = r + 8*halfid, N = mn
  const int pxg = pxbase + col;
#pragma unroll
  for (int r = 0; r < 8; ++r) {
    int dO = dotile * 16 + 8 * halfid + r;
    out[dO * NPIX + pxg] = acc[r];
  }
}

extern "C" void kernel_launch(void* const* d_in, const int* in_sizes, int n_in,
                              void* d_out, int out_size, void* d_ws, size_t ws_size,
                              hipStream_t stream) {
  (void)in_sizes; (void)n_in; (void)out_size; (void)ws_size;
  const float* incl   = (const float*)d_in[0];
  const float* rot    = (const float*)d_in[1];
  const float* lb     = (const float*)d_in[2];
  const float* vshift = (const float*)d_in[3];
  const float* vmax   = (const float*)d_in[4];
  const float* rturn  = (const float*)d_in[5];
  const float* i0     = (const float*)d_in[6];
  const float* rd     = (const float*)d_in[7];
  _Float16* T   = (_Float16*)d_ws;   // 64*256 f16 = 32 KB
  float*    out = (float*)d_out;     // 64*128*128 f32

  k_spectral_table<<<(DOUT * JQ + 255) / 256, 256, 0, stream>>>(incl, lb, vshift, vmax, T);
  k_main<<<NPIX / 32, 256, 0, stream>>>(incl, rot, lb, vmax, rturn, i0, rd, T, out);
}
